// HungarianMatcher_35897336660204
// MI455X (gfx1250) — matrix operand order repaired and checked by reading, not compiled, not run on hardware
//
#include <hip/hip_runtime.h>
#include <hip/hip_bf16.h>
#include <cstdint>

typedef __attribute__((ext_vector_type(16))) __bf16 v16bf;
typedef __attribute__((ext_vector_type(8)))  float  v8f;
typedef __attribute__((ext_vector_type(4)))  unsigned int v4u;
typedef __attribute__((ext_vector_type(8)))  int    v8i;
typedef __attribute__((ext_vector_type(4)))  int    v4i;

#define B_SZ  128
#define Q_SZ  900
#define C_SZ  256
#define N_SZ  32
#define QP    928      // padded Q stride for the cost scratch (32-aligned)
#define INF_F 1e30f

__device__ __forceinline__ float sigmoidf_(float x) {
    return 1.0f / (1.0f + __expf(-x));
}

// ---------------------------------------------------------------------------
// Kernel A: build transposed cost matrix costT[b][n][q] (f32, stride QP).
// Class cost via one-hot WMMA matmul: prob[16q x 32c] (bf16 hi+lo split)
// times onehot[32c x 16n] accumulated in f32.  Box L1 + GIoU fused in the
// epilogue using LDS-staged boxes.
// Block: 256 threads = 8 waves; each wave owns a 16-row q tile; grid (B, 8).
// ---------------------------------------------------------------------------
__global__ __launch_bounds__(256) void cost_build_kernel(
    const float* __restrict__ logits,   // [B,Q,C]
    const float* __restrict__ pboxes,   // [B,Q,4] cxcywh
    const int*   __restrict__ labels,   // [B,N]
    const float* __restrict__ tboxes,   // [B,N,4] cxcywh
    float*       __restrict__ costT)    // [B,N,QP]
{
    __shared__ float sPB[128][4];
    __shared__ float sTB[N_SZ][4];
    __shared__ int   sLab[N_SZ];

    const int b   = blockIdx.x;
    const int q0b = blockIdx.y * 128;
    const int tid = threadIdx.x;

    if (tid < 128) {
        int q  = q0b + tid;
        int qc = q < Q_SZ ? q : (Q_SZ - 1);
        const float4 p = *(const float4*)(pboxes + ((size_t)b * Q_SZ + qc) * 4);
        sPB[tid][0] = p.x; sPB[tid][1] = p.y; sPB[tid][2] = p.z; sPB[tid][3] = p.w;
    } else if (tid < 128 + N_SZ) {
        int n = tid - 128;
        const float4 t = *(const float4*)(tboxes + ((size_t)b * N_SZ + n) * 4);
        sTB[n][0] = t.x; sTB[n][1] = t.y; sTB[n][2] = t.z; sTB[n][3] = t.w;
    } else if (tid < 128 + 2 * N_SZ) {
        int n = tid - 128 - N_SZ;
        sLab[n] = labels[b * N_SZ + n];
    }
    __syncthreads();

    const int  wave    = tid >> 5;
    const int  lane    = tid & 31;
    const int  q0      = q0b + wave * 16;
    if (q0 >= Q_SZ) return;              // whole-wave uniform exit (EXEC stays all-1)

    const int  m       = lane & 15;
    const bool lo_half = lane < 16;
    const int  koff    = lo_half ? 0 : 8;   // A-fragment K sub-offset (ISA 16-bit A layout)
    const int  kb      = lo_half ? 0 : 16;  // unused; B fragment uses koff like A

    int qa = q0 + m; if (qa >= Q_SZ) qa = Q_SZ - 1;   // clamp tail reads
    const float* rowp = logits + ((size_t)b * Q_SZ + qa) * C_SZ;

    const int labn0 = sLab[m];
    const int labn1 = sLab[m + 16];

    v8f acc0 = {}; v8f acc1 = {};

    for (int c0 = 0; c0 < C_SZ; c0 += 32) {
        if (c0 + 32 < C_SZ) __builtin_prefetch(rowp + c0 + 32, 0, 3);  // global_prefetch_b8

        // 16 logits per lane, A layout: elems 0..7 -> K=koff..koff+7,
        //                               elems 8..15 -> K=16+koff..16+koff+7
        const float4 f0 = *(const float4*)(rowp + c0 + koff);
        const float4 f1 = *(const float4*)(rowp + c0 + koff + 4);
        const float4 f2 = *(const float4*)(rowp + c0 + 16 + koff);
        const float4 f3 = *(const float4*)(rowp + c0 + 16 + koff + 4);
        float s[16] = { sigmoidf_(f0.x), sigmoidf_(f0.y), sigmoidf_(f0.z), sigmoidf_(f0.w),
                        sigmoidf_(f1.x), sigmoidf_(f1.y), sigmoidf_(f1.z), sigmoidf_(f1.w),
                        sigmoidf_(f2.x), sigmoidf_(f2.y), sigmoidf_(f2.z), sigmoidf_(f2.w),
                        sigmoidf_(f3.x), sigmoidf_(f3.y), sigmoidf_(f3.z), sigmoidf_(f3.w) };

        v16bf ahi, alo;
        #pragma unroll
        for (int i = 0; i < 16; ++i) {
            float  sv = s[i];
            __bf16 h  = (__bf16)sv;
            ahi[i] = h;
            alo[i] = (__bf16)(sv - (float)h);   // residual -> ~16 mantissa bits total
        }

        // One-hot B fragments for the two 16-wide n tiles.
        v16bf b0, b1;
        #pragma unroll
        for (int j = 0; j < 16; ++j) {
            int kk = c0 + koff + j + (j >= 8 ? 8 : 0);
            b0[j] = (labn0 == kk) ? (__bf16)1.0f : (__bf16)0.0f;
            b1[j] = (labn1 == kk) ? (__bf16)1.0f : (__bf16)0.0f;
        }

        acc0 = __builtin_amdgcn_wmma_f32_16x16x32_bf16(false, ahi, false, b0, (short)0, acc0, false, false);
        acc0 = __builtin_amdgcn_wmma_f32_16x16x32_bf16(false, alo, false, b0, (short)0, acc0, false, false);
        acc1 = __builtin_amdgcn_wmma_f32_16x16x32_bf16(false, ahi, false, b1, (short)0, acc1, false, false);
        acc1 = __builtin_amdgcn_wmma_f32_16x16x32_bf16(false, alo, false, b1, (short)0, acc1, false, false);
    }

    // Epilogue: D layout -> lane column n = m (+16*nt), rows q0 + v + (lo?0:8)
    const int qbase = q0 + (lo_half ? 0 : 8);
    #pragma unroll
    for (int nt = 0; nt < 2; ++nt) {
        const int n = m + 16 * nt;
        const float tcx = sTB[n][0], tcy = sTB[n][1], tw = sTB[n][2], th = sTB[n][3];
        const float tx0 = tcx - 0.5f * tw, ty0 = tcy - 0.5f * th;
        const float tx1 = tcx + 0.5f * tw, ty1 = tcy + 0.5f * th;
        const float a2  = (tx1 - tx0) * (ty1 - ty0);
        v8f accv = nt ? acc1 : acc0;

        #pragma unroll
        for (int v = 0; v < 8; ++v) {
            const int q = qbase + v;
            if (q >= Q_SZ) continue;
            const int ql = q - q0b;
            const float pcx = sPB[ql][0], pcy = sPB[ql][1], pw = sPB[ql][2], ph = sPB[ql][3];
            const float px0 = pcx - 0.5f * pw, py0 = pcy - 0.5f * ph;
            const float px1 = pcx + 0.5f * pw, py1 = pcy + 0.5f * ph;
            const float a1  = (px1 - px0) * (py1 - py0);

            const float iw    = fmaxf(fminf(px1, tx1) - fmaxf(px0, tx0), 0.0f);
            const float ih    = fmaxf(fminf(py1, ty1) - fmaxf(py0, ty0), 0.0f);
            const float inter = iw * ih;
            const float uni   = a1 + a2 - inter;
            const float ew    = fmaxf(fmaxf(px1, tx1) - fminf(px0, tx0), 0.0f);
            const float eh    = fmaxf(fmaxf(py1, ty1) - fminf(py0, ty0), 0.0f);
            const float enc   = ew * eh;
            const float iou   = inter / fmaxf(uni, 1e-6f);
            const float giou  = iou - (enc - uni) / fmaxf(enc, 1e-6f);
            const float l1    = fabsf(pcx - tcx) + fabsf(pcy - tcy) +
                                fabsf(pw - tw) + fabsf(ph - th);
            const float prob  = accv[v];
            costT[((size_t)b * N_SZ + n) * QP + q] = -2.0f * prob + 5.0f * l1 - 2.0f * giou;
        }
    }
}

// ---------------------------------------------------------------------------
// Kernel B: Jonker-Volgenant LSA per batch, fully LDS-resident (~131 KB).
// Cost block staged by the Tensor Data Mover (tensor_load_to_lds) while the
// other waves initialize duals; fused rank-sort output.
// One 256-thread block per batch.
// ---------------------------------------------------------------------------
__global__ __launch_bounds__(256) void lsa_kernel(
    const float* __restrict__ costT,   // [B,N,QP]
    int* __restrict__ outRow,          // [B,N] query indices (sorted)
    int* __restrict__ outCol)          // [B,N] target indices
{
    __shared__ float costL[N_SZ * Q_SZ];   // 115200 B — needs CDNA5's big LDS
    __shared__ float shortestL[Q_SZ];
    __shared__ float vpotL[Q_SZ];
    __shared__ int   pathL[Q_SZ];
    __shared__ int   row4colL[Q_SZ];
    __shared__ unsigned char remL[Q_SZ];
    __shared__ float uL[N_SZ];
    __shared__ int   col4rowL[N_SZ];
    __shared__ unsigned char SRL[N_SZ];
    __shared__ float redV[8];
    __shared__ int   redI[8];
    __shared__ int   sh_i, sh_done, sh_sink;
    __shared__ float sh_minVal;

    const int b    = blockIdx.x;
    const int tid  = threadIdx.x;
    const int lane = tid & 31;
    const int wave = tid >> 5;

    // --- Stage cost via TDM (wave 0 issues the DMA; TDM ignores EXEC) ------
    if (tid < 32) {
        const float* src = costT + (size_t)b * N_SZ * QP;
        unsigned long long ga = (unsigned long long)(uintptr_t)src;
        // raw 32-bit LDS byte offset of costL (addrspace(3) cast, not flat)
        unsigned ldsa = (unsigned)(unsigned long long)
                        (__attribute__((address_space(3))) const void*)costL;

        v4u g0;
        g0[0] = 1u;                                    // count=1, user desc, no gather
        g0[1] = ldsa;                                  // lds_addr (bytes)
        g0[2] = (unsigned)(ga & 0xffffffffu);          // global_addr[31:0]
        g0[3] = (unsigned)((ga >> 32) & 0x01ffffffu)   // global_addr[56:32]
                | (2u << 30);                          // type=2 ("image")

        const unsigned td0 = QP, td1 = N_SZ;           // tensor dims (elements)
        const unsigned ti0 = Q_SZ, ti1 = N_SZ;         // tile dims
        const unsigned long long st0 = QP;             // dim0 stride (elements)
        v8i g1;
        g1[0] = (int)(2u << 16);                       // data_size=2 (4 bytes)
        g1[1] = (int)((td0 & 0xffffu) << 16);          // tensor_dim0[15:0]
        g1[2] = (int)(((td0 >> 16) & 0xffffu) | ((td1 & 0xffffu) << 16));
        g1[3] = (int)(((td1 >> 16) & 0xffffu) | ((ti0 & 0xffffu) << 16)); // tile_dim0
        g1[4] = (int)(ti1 & 0xffffu);                  // tile_dim1; tile_dim2=0
        g1[5] = (int)(st0 & 0xffffffffu);              // tensor_dim0_stride[31:0]
        g1[6] = (int)((st0 >> 32) & 0xffffu);          // stride0[47:32]; stride1[15:0]=0
        g1[7] = 0;

        v4i g2 = {}; v4i g3 = {};                      // dims 2+ unused (2D tile)
#if defined(__clang_major__) && (__clang_major__ >= 23)
        v8i g4 = {};
        __builtin_amdgcn_tensor_load_to_lds(g0, g1, g2, g3, g4, 0);
#else
        __builtin_amdgcn_tensor_load_to_lds(g0, g1, g2, g3, 0);
#endif
        __builtin_amdgcn_s_wait_tensorcnt(0);          // s_wait_tensorcnt 0
    }
    // overlap: dual-potential / assignment init while the DMA runs
    for (int q = tid; q < Q_SZ; q += 256) { vpotL[q] = 0.0f; row4colL[q] = -1; }
    if (tid < N_SZ) { uL[tid] = 0.0f; col4rowL[tid] = -1; }
    __syncthreads();

    for (int cur_row = 0; cur_row < N_SZ; ++cur_row) {
        // per-row init
        for (int q = tid; q < Q_SZ; q += 256) {
            shortestL[q] = INF_F; pathL[q] = -1; remL[q] = 1;
        }
        if (tid < N_SZ) SRL[tid] = 0;
        if (tid == 0) { sh_i = cur_row; sh_minVal = 0.0f; sh_done = 0; }
        __syncthreads();

        // Dijkstra-style shortest augmenting path
        while (true) {
            if (tid == 0) SRL[sh_i] = 1;
            const int   i  = sh_i;
            const float mv = sh_minVal;
            const float ui = uL[i];
            for (int q = tid; q < Q_SZ; q += 256) {
                if (remL[q]) {
                    float cand = mv + costL[i * Q_SZ + q] - ui - vpotL[q];
                    if (cand < shortestL[q]) { shortestL[q] = cand; pathL[q] = i; }
                }
            }
            __syncthreads();

            // argmin over remaining (first-min tie-break): wave shuffle + tiny LDS combine
            float bv = 3.0e38f; int bi = 0x7fffffff;
            for (int q = tid; q < Q_SZ; q += 256) {
                if (remL[q]) {
                    float sv = shortestL[q];
                    if (sv < bv || (sv == bv && q < bi)) { bv = sv; bi = q; }
                }
            }
            #pragma unroll
            for (int msk = 16; msk > 0; msk >>= 1) {
                float ov = __shfl_xor(bv, msk, 32);
                int   oi = __shfl_xor(bi, msk, 32);
                if (ov < bv || (ov == bv && oi < bi)) { bv = ov; bi = oi; }
            }
            if (lane == 0) { redV[wave] = bv; redI[wave] = bi; }
            __syncthreads();

            if (tid == 0) {
                float fv = redV[0]; int fj = redI[0];
                #pragma unroll
                for (int w = 1; w < 8; ++w) {
                    float ov = redV[w]; int oi = redI[w];
                    if (ov < fv || (ov == fv && oi < fj)) { fv = ov; fj = oi; }
                }
                remL[fj]   = 0;
                sh_minVal  = fv;
                int r = row4colL[fj];
                if (r < 0) { sh_sink = fj; sh_done = 1; }
                else       { sh_i = r; }
            }
            __syncthreads();
            if (sh_done) break;
        }

        // dual potential updates
        if (tid < N_SZ && SRL[tid]) {
            int   cr = col4rowL[tid] < 0 ? 0 : col4rowL[tid];
            float du = (tid == cur_row) ? sh_minVal : (sh_minVal - shortestL[cr]);
            uL[tid] += du;
        }
        for (int q = tid; q < Q_SZ; q += 256)
            if (!remL[q]) vpotL[q] -= sh_minVal - shortestL[q];
        __syncthreads();

        // augment along path (<=32 steps, serial)
        if (tid == 0) {
            int sink = sh_sink;
            while (true) {
                int i = pathL[sink];
                row4colL[sink] = i;
                int ns = col4rowL[i];
                col4rowL[i] = sink;
                if (i == cur_row) break;
                sink = ns;
            }
        }
        __syncthreads();
    }

    // fused output: sort by query id via rank (values are distinct)
    if (tid < N_SZ) {
        int c = col4rowL[tid];
        int rank = 0;
        #pragma unroll
        for (int mm = 0; mm < N_SZ; ++mm) rank += (col4rowL[mm] < c) ? 1 : 0;
        outRow[b * N_SZ + rank] = c;
        outCol[b * N_SZ + rank] = tid;
    }
}

extern "C" void kernel_launch(void* const* d_in, const int* in_sizes, int n_in,
                              void* d_out, int out_size, void* d_ws, size_t ws_size,
                              hipStream_t stream) {
    const float* logits = (const float*)d_in[0];   // [128,900,256] f32
    const float* pboxes = (const float*)d_in[1];   // [128,900,4]   f32
    const int*   labels = (const int*)  d_in[2];   // [128,32]      i32
    const float* tboxes = (const float*)d_in[3];   // [128,32,4]    f32

    float* costT = (float*)d_ws;                   // needs 128*32*928*4 = ~14.5 MB
    int*   out   = (int*)d_out;                    // [row_idx | col_idx], 2*128*32 i32

    dim3 gA(B_SZ, (Q_SZ + 127) / 128);
    cost_build_kernel<<<gA, 256, 0, stream>>>(logits, pboxes, labels, tboxes, costT);
    lsa_kernel<<<B_SZ, 256, 0, stream>>>(costT, out, out + B_SZ * N_SZ);
}
